// T2ICrossAttentionCoding_70231305224557
// MI455X (gfx1250) — matrix operand order repaired and checked by reading, not compiled, not run on hardware
//
#include <hip/hip_runtime.h>
#include <hip/hip_bf16.h>
#include <math.h>

// Problem sizes (compile-time constants from the reference)
#define I_N 64
#define R_N 36
#define R_P 48      // regions padded to 3x16 WMMA tiles (rows 36..47 zero)
#define D_N 1024
#define C_N 64
#define W_N 32
#define SMOOTH_F 9.0f
#define EPS_F 1e-8f

typedef __attribute__((ext_vector_type(16))) __bf16 v16bf;
typedef __attribute__((ext_vector_type(8)))  float  v8f;

union FragU { v16bf v; uint4 q[2]; };

__device__ __forceinline__ unsigned short f2bf(float f) {
  unsigned u = __float_as_uint(f);
  unsigned r = u + 0x7FFFu + ((u >> 16) & 1u);   // RNE
  return (unsigned short)(r >> 16);
}

// A-matrix fragment, 16x32 bf16, row-major source with row stride D_N elems.
// ISA layout: lane m=lane&15; VGPR0-3 hold K = kb..kb+7, VGPR4-7 hold K = kb+16..kb+23,
// where kb = 8*(lane>=16).  => two 16B loads at (m, kb) and (m, kb+16).
__device__ __forceinline__ v16bf load_fragA(const unsigned short* base, int lane) {
  int m = lane & 15, kh = (lane >> 4) * 8;
  const uint4* p = (const uint4*)(base + (size_t)m * D_N + kh);
  FragU f; f.q[0] = p[0]; f.q[1] = p[2];        // +0 elems, +16 elems
  return f.v;
}
// B-matrix fragment, 32x16 bf16, built from B^T rows (row-major, stride D_N):
// lane n=lane&15; VGPR0-3 hold K = kb..kb+7, VGPR4-7 hold K = kb+16..kb+23, kb = 8*(lane>=16).
__device__ __forceinline__ v16bf load_fragB(const unsigned short* base, int lane) {
  int n = lane & 15, kh = (lane >> 4) * 8;
  const uint4* p = (const uint4*)(base + (size_t)n * D_N + kh);
  FragU f; f.q[0] = p[0]; f.q[1] = p[2];        // +0 elems, +16 elems
  return f.v;
}

// ---------------- precompute kernels ----------------

__global__ void convert_imgs_kernel(const float* __restrict__ imgs,
                                    unsigned short* __restrict__ out) {
  int idx = blockIdx.x * 256 + threadIdx.x;
  if (idx >= I_N * R_P * D_N) return;
  int d = idx & (D_N - 1);
  int r = (idx >> 10) % R_P;
  int i = idx / (R_P * D_N);
  out[idx] = (r < R_N) ? f2bf(imgs[((size_t)i * R_N + r) * D_N + d])
                       : (unsigned short)0;
}

__global__ void convert_caps_kernel(const float* __restrict__ caps,
                                    unsigned short* __restrict__ out) {
  int idx = blockIdx.x * 256 + threadIdx.x;
  if (idx < C_N * W_N * D_N) out[idx] = f2bf(caps[idx]);
}

__global__ void __launch_bounds__(32)
capnorm_kernel(const float* __restrict__ caps, float* __restrict__ capnorm) {
  int b = blockIdx.x;                   // c*W + w
  int lane = threadIdx.x;
  const float* row = caps + (size_t)b * D_N;
  float s = 0.f;
  for (int d = lane; d < D_N; d += 32) { float v = row[d]; s += v * v; }
  for (int m = 1; m < 32; m <<= 1) s += __shfl_xor(s, m, 32);
  if (lane == 0) capnorm[b] = sqrtf(s);
}

// G_i = imgs_i * imgs_i^T  (48x48 padded, bf16 WMMA, f32 accumulate)
__global__ void __launch_bounds__(32)
gram_kernel(const unsigned short* __restrict__ imgs_bf, float* __restrict__ G) {
  int i = blockIdx.x, lane = threadIdx.x;
  const unsigned short* P = imgs_bf + (size_t)i * R_P * D_N;
  v8f zero = {0.f, 0.f, 0.f, 0.f, 0.f, 0.f, 0.f, 0.f};
  v8f acc[3][3];
  for (int a = 0; a < 3; a++)
    for (int b = 0; b < 3; b++) acc[a][b] = zero;
  for (int k = 0; k < D_N; k += 32) {
    v16bf af[3], bfm[3];
    for (int t = 0; t < 3; t++) af[t]  = load_fragA(P + (size_t)t * 16 * D_N + k, lane);
    for (int t = 0; t < 3; t++) bfm[t] = load_fragB(P + (size_t)t * 16 * D_N + k, lane);
    for (int mt = 0; mt < 3; mt++)
      for (int nt = 0; nt < 3; nt++)
        acc[mt][nt] = __builtin_amdgcn_wmma_f32_16x16x32_bf16(
            false, af[mt], false, bfm[nt], (short)0, acc[mt][nt], false, false);
  }
  int hi = lane >> 4, n = lane & 15;
  float* Gi = G + (size_t)i * R_P * R_P;
  for (int mt = 0; mt < 3; mt++)
    for (int nt = 0; nt < 3; nt++)
      for (int j = 0; j < 8; j++)
        Gi[(mt * 16 + hi * 8 + j) * R_P + nt * 16 + n] = acc[mt][nt][j];
}

// ---------------- fused main kernel: one wave32 per (c,i) ----------------

__global__ void __launch_bounds__(32)
fused_kernel(const unsigned short* __restrict__ imgs_bf,
             const unsigned short* __restrict__ caps_bf,
             const float* __restrict__ G,
             const float* __restrict__ capnorm,
             const int* __restrict__ cap_lens,
             float* __restrict__ out) {
  __shared__ float s_attn[W_N][R_P + 1];   // stride 49 -> conflict-free column walks
  __shared__ float s_g[R_P * R_P];
  __shared__ float s_w12[W_N];

  int bid = blockIdx.x;
  int c = bid >> 6;        // / I_N
  int i = bid & 63;
  int lane = threadIdx.x;
  int hi = lane >> 4, n = lane & 15;

  const unsigned short* A = caps_bf + (size_t)c * W_N * D_N;  // 32 x 1024
  const unsigned short* B = imgs_bf + (size_t)i * R_P * D_N;  // 48 x 1024 (padded)

  // stage Gram matrix into LDS (2304 floats = 576 uint4)
  {
    const uint4* gs = (const uint4*)(G + (size_t)i * R_P * R_P);
    uint4* gd = (uint4*)s_g;
    for (int t = lane; t < (R_P * R_P) / 4; t += 32) gd[t] = gs[t];
  }

  // Phase A: S[w,r] = cap_w . img_r   (2x3 tiles of 16x16, K=1024)
  v8f zero = {0.f, 0.f, 0.f, 0.f, 0.f, 0.f, 0.f, 0.f};
  v8f S[2][3];
  for (int a = 0; a < 2; a++)
    for (int b = 0; b < 3; b++) S[a][b] = zero;
  for (int k = 0; k < D_N; k += 32) {
    v16bf af[2], bfm[3];
    af[0] = load_fragA(A + k, lane);
    af[1] = load_fragA(A + (size_t)16 * D_N + k, lane);
    for (int t = 0; t < 3; t++) bfm[t] = load_fragB(B + (size_t)t * 16 * D_N + k, lane);
    for (int mt = 0; mt < 2; mt++)
      for (int nt = 0; nt < 3; nt++)
        S[mt][nt] = __builtin_amdgcn_wmma_f32_16x16x32_bf16(
            false, af[mt], false, bfm[nt], (short)0, S[mt][nt], false, false);
  }

  int caplen = cap_lens[c];

  // Phase B1: leaky_relu(0.1) then word mask  (element w = mt*16 + 8*hi + j, r = nt*16 + n)
  v8f At[2][3];
  for (int mt = 0; mt < 2; mt++)
    for (int nt = 0; nt < 3; nt++)
      for (int j = 0; j < 8; j++) {
        float s = S[mt][nt][j];
        float a = s > 0.f ? s : 0.1f * s;
        int w = mt * 16 + hi * 8 + j;
        At[mt][nt][j] = (w < caplen) ? a : 0.f;
      }

  // Phase B2: L2 normalize over w for each region column r
  for (int nt = 0; nt < 3; nt++) {
    float ss = 0.f;
    for (int mt = 0; mt < 2; mt++)
      for (int j = 0; j < 8; j++) ss += At[mt][nt][j] * At[mt][nt][j];
    ss += __shfl_xor(ss, 16, 32);                 // lane L and L^16 share r
    float inv = 1.f / (sqrtf(ss) + EPS_F);
    for (int mt = 0; mt < 2; mt++)
      for (int j = 0; j < 8; j++) At[mt][nt][j] *= inv;
  }

  // Phase B3: softmax over r (valid r < 36), logits = SMOOTH * a
  for (int mt = 0; mt < 2; mt++)
    for (int j = 0; j < 8; j++) {
      float l[3], mx = -3.4e38f;
      for (int nt = 0; nt < 3; nt++) {
        int r = nt * 16 + n;
        l[nt] = (r < R_N) ? SMOOTH_F * At[mt][nt][j] : -3.4e38f;
        mx = fmaxf(mx, l[nt]);
      }
      for (int m = 1; m < 16; m <<= 1) mx = fmaxf(mx, __shfl_xor(mx, m, 32));
      float e[3], se = 0.f;
      for (int nt = 0; nt < 3; nt++) {
        e[nt] = (l[nt] > -1e37f) ? expf(l[nt] - mx) : 0.f;
        se += e[nt];
      }
      for (int m = 1; m < 16; m <<= 1) se += __shfl_xor(se, m, 32);
      float isum = 1.f / se;
      for (int nt = 0; nt < 3; nt++) At[mt][nt][j] = e[nt] * isum;
    }

  // Phase B4: w12[w] = sum_r attn_final * S_raw  (== dot(cap_w, wctx_w) unnormalized)
  for (int mt = 0; mt < 2; mt++)
    for (int j = 0; j < 8; j++) {
      float p = 0.f;
      for (int nt = 0; nt < 3; nt++) p += At[mt][nt][j] * S[mt][nt][j];
      for (int m = 1; m < 16; m <<= 1) p += __shfl_xor(p, m, 32);
      int w = mt * 16 + hi * 8 + j;
      if (n == 0) s_w12[w] = p;
      for (int nt = 0; nt < 3; nt++)
        s_attn[w][nt * 16 + n] = At[mt][nt][j];
    }
  __syncthreads();

  // Phase C: ||wctx_w||^2 = attn_w^T G attn_w ; cosine sim; mask; store
  int w = lane;
  float av[R_N];
  for (int r = 0; r < R_N; r++) av[r] = s_attn[w][r];
  float q = 0.f;
  for (int rp = 0; rp < R_N; rp++) {
    float t = 0.f;
    for (int r = 0; r < R_N; r++) t += av[r] * s_g[r * R_P + rp];  // broadcast reads
    q += av[rp] * t;
  }
  float nv = sqrtf(fmaxf(q, 0.f));
  float w12 = s_w12[w] / (nv + EPS_F);       // dot(cap, wctx/(||wctx||+eps))
  float w2 = nv / (nv + EPS_F);              // || normalized wctx ||
  float w1 = capnorm[c * W_N + w];
  float sim = w12 / fmaxf(w1 * w2, EPS_F);
  if (w >= caplen) sim = -1.0f;
  out[((size_t)i * C_N + c) * W_N + w] = sim;  // output (I, C, W)
}

// ---------------- launch ----------------

extern "C" void kernel_launch(void* const* d_in, const int* in_sizes, int n_in,
                              void* d_out, int out_size, void* d_ws, size_t ws_size,
                              hipStream_t stream) {
  const float* imgs = (const float*)d_in[0];
  const float* caps = (const float*)d_in[1];
  // d_in[2] = img_lens: unused by the reference math
  const int* cap_lens = (const int*)d_in[3];
  float* out = (float*)d_out;

  // workspace layout (~11.1 MB total)
  char* ws = (char*)d_ws;
  unsigned short* ws_imgs = (unsigned short*)ws;                       // 64*48*1024*2 = 6,291,456 B
  unsigned short* ws_caps = (unsigned short*)(ws + 6291456);           // 64*32*1024*2 = 4,194,304 B
  float* ws_G    = (float*)(ws + 6291456 + 4194304);                   // 64*48*48*4   =   589,824 B
  float* ws_norm = (float*)(ws + 6291456 + 4194304 + 589824);          // 64*32*4      =     8,192 B

  convert_imgs_kernel<<<(I_N * R_P * D_N + 255) / 256, 256, 0, stream>>>(imgs, ws_imgs);
  convert_caps_kernel<<<(C_N * W_N * D_N + 255) / 256, 256, 0, stream>>>(caps, ws_caps);
  capnorm_kernel<<<C_N * W_N, 32, 0, stream>>>(caps, ws_norm);
  gram_kernel<<<I_N, 32, 0, stream>>>(ws_imgs, ws_G);
  fused_kernel<<<C_N * I_N, 32, 0, stream>>>(ws_imgs, ws_caps, ws_G, ws_norm,
                                             cap_lens, out);
}
